// DiscreteSSL_ESC50_7816840479208
// MI455X (gfx1250) — matrix operand order repaired and checked by reading, not compiled, not run on hardware
//
#include <hip/hip_runtime.h>
#include <hip/hip_bf16.h>

// Problem constants
#define LL 4
#define BB 16
#define DD 768
#define SS 1000          // T*F
#define KK 1000          // clusters
#define KP 1024          // padded clusters
#define KT 512           // target dim
#define NSB 32           // ceil(S/32) s-blocks of 32 rows
#define NDC (DD / 32)    // 24 d-chunks of 32

typedef __attribute__((ext_vector_type(16))) __bf16 v16bf;
typedef __attribute__((ext_vector_type(8)))  float  v8f;

typedef unsigned short u16;
typedef unsigned int   u32;

// ---------- helpers ----------
__device__ __forceinline__ u16 f2bf(float x) {
    u32 u = __builtin_bit_cast(u32, x);
    u32 r = u + 0x7FFFu + ((u >> 16) & 1u);   // round-to-nearest-even
    return (u16)(r >> 16);
}

__device__ __forceinline__ int imin(int a, int b) { return a < b ? a : b; }

// A fragment (16x32 bf16) from LDS, row-major [row][d], rowStride even.
__device__ __forceinline__ v16bf load_a_frag(const u16* lds, int rowStride,
                                             int mbase, int kbase) {
    int lane = threadIdx.x & 31;
    int m    = lane & 15;
    int half = lane >> 4;
    const u16* p = lds + (mbase + m) * rowStride + kbase + half * 16;
    union { v16bf v; u32 u[8]; } f;
    f.u[0] = *(const u32*)(p + 0);
    f.u[1] = *(const u32*)(p + 2);
    f.u[2] = *(const u32*)(p + 4);
    f.u[3] = *(const u32*)(p + 6);
    f.u[4] = *(const u32*)(p + 8);
    f.u[5] = *(const u32*)(p + 10);
    f.u[6] = *(const u32*)(p + 12);
    f.u[7] = *(const u32*)(p + 14);
    return f.v;
}

// B fragment pre-swizzled in global: [lane][16 bf16], two b128 per lane.
__device__ __forceinline__ v16bf load_b_frag_g(const u16* g) {
    int lane = threadIdx.x & 31;
    const uint4* p = (const uint4*)(g + lane * 16);
    union { v16bf v; uint4 q[2]; } f;
    f.q[0] = p[0];
    f.q[1] = p[1];
    return f.v;
}

__device__ __forceinline__ v8f wmma_bf16(v16bf a, v16bf b, v8f c) {
    return __builtin_amdgcn_wmma_f32_16x16x32_bf16(false, a, false, b,
                                                   (short)0, c, false, false);
}

// ---------- kernel 1: centers -> bf16 rows (padded) + c2 ----------
__global__ void k_prep_centers(const float* __restrict__ centers,
                               u16* __restrict__ cb16,
                               float* __restrict__ c2w) {
    int l = blockIdx.x / KP;
    int k = blockIdx.x % KP;
    int t = threadIdx.x;
    __shared__ float warp_s[8];
    u16* dst = cb16 + (size_t)(l * KP + k) * DD;
    float part = 0.f;
    if (k < KK) {
        const float* src = centers + (size_t)(l * KK + k) * DD;
        for (int j = t; j < DD; j += 256) {
            float v = src[j];
            dst[j]  = f2bf(v);
            part += v * v;
        }
    } else {
        for (int j = t; j < DD; j += 256) dst[j] = 0;
    }
#pragma unroll
    for (int off = 16; off >= 1; off >>= 1) part += __shfl_xor(part, off, 32);
    if ((t & 31) == 0) warp_s[t >> 5] = part;
    __syncthreads();
    if (t == 0) {
        float s = 0.f;
#pragma unroll
        for (int w = 0; w < 8; ++w) s += warp_s[w];
        c2w[l * KP + k] = (k < KK) ? s : 3.0e38f;
    }
}

// ---------- kernel 2: pack centers into WMMA B-fragment order ----------
// cpack[((l*64 + ktile)*NDC + dc)*512 + lane*16 + 2i] holds the bf16 pair
// (cluster = ktile*16 + (lane&15),  k = dc*32 + (lane>>4)*16 + 2i)
__global__ void k_pack_centers(const u16* __restrict__ cb16,
                               u32* __restrict__ cpack) {
    int p = blockIdx.x * 256 + threadIdx.x;     // L*64*NDC*256 pairs
    int i     = p & 7;
    int lane  = (p >> 3) & 31;
    int g     = p >> 8;                         // (l,ktile,dc)
    int dc    = g % NDC;
    int ktile = (g / NDC) % 64;
    int l     = g / (NDC * 64);
    int cluster = ktile * 16 + (lane & 15);
    int k       = dc * 32 + (lane >> 4) * 16 + 2 * i;
    cpack[p] = *(const u32*)(cb16 + (size_t)(l * KP + cluster) * DD + k);
}

// ---------- kernel 3: pack W into WMMA B-fragment order ----------
// wpack[((l*32 + kttile)*NDC + dc)*512 + lane*16 + 2i] = (W[k][kt], W[k+1][kt])
__global__ void k_pack_w(const float* __restrict__ W, u32* __restrict__ wpack) {
    int p = blockIdx.x * 256 + threadIdx.x;     // L*32*NDC*256 pairs
    int i      = p & 7;
    int lane   = (p >> 3) & 31;
    int g      = p >> 8;
    int dc     = g % NDC;
    int kttile = (g / NDC) % 32;
    int l      = g / (NDC * 32);
    int kt = kttile * 16 + (lane & 15);
    int k  = dc * 32 + (lane >> 4) * 16 + 2 * i;
    float f0 = W[(size_t)(l * DD + k) * KT + kt];
    float f1 = W[(size_t)(l * DD + k + 1) * KT + kt];
    wpack[p] = (u32)f2bf(f0) | ((u32)f2bf(f1) << 16);
}

// ---------- kernel 4: distance GEMM + argmin -> tokens ----------
__global__ void k_tokens(const float* __restrict__ h,
                         const u16* __restrict__ cpack,
                         const float* __restrict__ c2w,
                         int* __restrict__ toks,
                         float* __restrict__ tok_out) {
    int sb = blockIdx.x % NSB;
    int bb = (blockIdx.x / NSB) % BB;
    int l  = blockIdx.x / (NSB * BB);
    int t  = threadIdx.x;
    int lane = t & 31;
    int wid  = t >> 5;
    int mt  = wid & 1;          // 2 m-tiles of 16 rows
    int ntq = wid >> 1;         // 4 wave-columns, each owning 2 n-tiles

    __shared__ u16 a_lds[32][DD + 8];      // 49,664 B bf16 X-rows
    __shared__ float bestv_s[32][8];
    __shared__ int   besti_s[32][8];

    // --- stage 32 S-rows of X (strided view of h) as bf16, once ---
    int sbase = sb * 32;
    {
        int s_local = t & 31;
        int dgrp    = t >> 5;
        int s = imin(sbase + s_local, SS - 1);
        const float* xb = h + ((size_t)(l * BB + bb) * DD) * SS + s;
        for (int d = dgrp; d < DD; d += 8)
            a_lds[s_local][d] = f2bf(xb[(size_t)d * SS]);
    }
    __syncthreads();   // the only barrier before the reduction

    float best[2][8];
    int   bidx[2][8];
#pragma unroll
    for (int j = 0; j < 2; ++j)
#pragma unroll
        for (int r = 0; r < 8; ++r) { best[j][r] = 3.3e38f; bidx[j][r] = 0; }

    // --- 8 blocks of 128 clusters; no barriers, no LDS for B ---
    for (int nb = 0; nb < KP / 128; ++nb) {
        int ktile0 = nb * 8 + ntq * 2;
        const u16* b0base = cpack + ((size_t)(l * 64 + ktile0) * NDC) * 512;
        const u16* b1base = b0base + (size_t)NDC * 512;
        v8f acc[2];
        acc[0] = (v8f){0.f,0.f,0.f,0.f,0.f,0.f,0.f,0.f};
        acc[1] = (v8f){0.f,0.f,0.f,0.f,0.f,0.f,0.f,0.f};
#pragma unroll 4
        for (int dc = 0; dc < NDC; ++dc) {
            v16bf a  = load_a_frag(&a_lds[0][0], DD + 8, mt * 16, dc * 32);
            v16bf b0 = load_b_frag_g(b0base + dc * 512);
            v16bf b1 = load_b_frag_g(b1base + dc * 512);
            acc[0] = wmma_bf16(a, b0, acc[0]);
            acc[1] = wmma_bf16(a, b1, acc[1]);
        }
        // running argmin:  dist = c2[k] - 2*G   (x2 constant per row)
#pragma unroll
        for (int j = 0; j < 2; ++j) {
            int   kg  = (ktile0 + j) * 16 + (lane & 15);
            float c2l = c2w[l * KP + kg];
#pragma unroll
            for (int r = 0; r < 8; ++r) {
                float val = c2l - 2.0f * acc[j][r];
                if (val < best[j][r]) { best[j][r] = val; bidx[j][r] = kg; }
            }
        }
    }

    // --- reduce across the 16 lanes of each half (ties -> lowest index) ---
#pragma unroll
    for (int j = 0; j < 2; ++j)
#pragma unroll
    for (int r = 0; r < 8; ++r) {
#pragma unroll
        for (int off = 8; off >= 1; off >>= 1) {
            float ov = __shfl_xor(best[j][r], off, 32);
            int   oi = __shfl_xor(bidx[j][r], off, 32);
            if (ov < best[j][r] || (ov == best[j][r] && oi < bidx[j][r])) {
                best[j][r] = ov; bidx[j][r] = oi;
            }
        }
        if ((lane & 15) == 0) {
            int m_loc = mt * 16 + (lane >> 4) * 8 + r;
            bestv_s[m_loc][ntq * 2 + j] = best[j][r];
            besti_s[m_loc][ntq * 2 + j] = bidx[j][r];
        }
    }
    __syncthreads();

    // --- final reduce across the 8 n-tile columns, write tokens ---
    if (t < 32) {
        float v = bestv_s[t][0];
        int   i = besti_s[t][0];
#pragma unroll
        for (int c = 1; c < 8; ++c) {
            float ov = bestv_s[t][c];
            int   oi = besti_s[t][c];
            if (ov < v || (ov == v && oi < i)) { v = ov; i = oi; }
        }
        int s = sbase + t;
        if (s < SS) {
            toks[(size_t)(l * BB + bb) * SS + s] = i;
            tok_out[(size_t)(bb * SS + s) * LL + l] = (float)i;
        }
    }
}

// ---------- kernel 5: embedding lookup + projection GEMM ----------
__global__ void k_proj(const u16* __restrict__ cb16,
                       const u16* __restrict__ wpack,
                       const float* __restrict__ bias,
                       const int* __restrict__ toks,
                       float* __restrict__ emb_out) {
    int sb = blockIdx.x % NSB;
    int bb = (blockIdx.x / NSB) % BB;
    int l  = blockIdx.x / (NSB * BB);
    int t  = threadIdx.x;
    int lane = t & 31;
    int wid  = t >> 5;
    int mt  = wid & 1;
    int ntq = wid >> 1;

    __shared__ u16 a_lds[32][DD + 8];   // gathered bf16 center rows

    int sbase = sb * 32;
    {   // gather: 8 threads per row, pure 16B bf16 copies
        int row = t >> 3;
        int seg = t & 7;
        int s   = imin(sbase + row, SS - 1);
        int tok = toks[(size_t)(l * BB + bb) * SS + s];
        const u16* src = cb16 + (size_t)(l * KP + tok) * DD + seg * 96;
#pragma unroll
        for (int j = 0; j < 12; ++j)
            *(uint4*)&a_lds[row][seg * 96 + j * 8] = *(const uint4*)(src + j * 8);
    }
    __syncthreads();   // only barrier

    // 4 blocks of 128 kt; B fragments straight from global, no barriers
    for (int nkt = 0; nkt < KT / 128; ++nkt) {
        int kttile0 = nkt * 8 + ntq * 2;
        const u16* b0base = wpack + ((size_t)(l * 32 + kttile0) * NDC) * 512;
        const u16* b1base = b0base + (size_t)NDC * 512;
        v8f acc[2];
        acc[0] = (v8f){0.f,0.f,0.f,0.f,0.f,0.f,0.f,0.f};
        acc[1] = (v8f){0.f,0.f,0.f,0.f,0.f,0.f,0.f,0.f};
#pragma unroll 4
        for (int dc = 0; dc < NDC; ++dc) {
            v16bf a  = load_a_frag(&a_lds[0][0], DD + 8, mt * 16, dc * 32);
            v16bf b0 = load_b_frag_g(b0base + dc * 512);
            v16bf b1 = load_b_frag_g(b1base + dc * 512);
            acc[0] = wmma_bf16(a, b0, acc[0]);
            acc[1] = wmma_bf16(a, b1, acc[1]);
        }
        // epilogue: add bias, store to [B,S,L,KT]
#pragma unroll
        for (int j = 0; j < 2; ++j) {
            int kt  = (kttile0 + j) * 16 + (lane & 15);
            float bv = bias[l * KT + kt];
#pragma unroll
            for (int r = 0; r < 8; ++r) {
                int m_loc = mt * 16 + (lane >> 4) * 8 + r;
                int s = sbase + m_loc;
                if (s < SS)
                    emb_out[((size_t)(bb * SS + s) * LL + l) * KT + kt] =
                        acc[j][r] + bv;
            }
        }
    }
}

// ---------- host launcher ----------
extern "C" void kernel_launch(void* const* d_in, const int* in_sizes, int n_in,
                              void* d_out, int out_size, void* d_ws, size_t ws_size,
                              hipStream_t stream) {
    const float* h       = (const float*)d_in[0];
    const float* centers = (const float*)d_in[1];
    const float* W       = (const float*)d_in[2];
    const float* bvec    = (const float*)d_in[3];

    // workspace layout
    const size_t sz_cb16  = (size_t)LL * KP * DD * sizeof(u16);        // 6,291,456
    const size_t sz_c2    = (size_t)LL * KP * sizeof(float);           //    16,384
    const size_t sz_cpack = (size_t)LL * 64 * NDC * 512 * sizeof(u16); // 6,291,456
    const size_t sz_wpack = (size_t)LL * 32 * NDC * 512 * sizeof(u16); // 3,145,728

    char* ws = (char*)d_ws;
    u16*   cb16  = (u16*)ws;
    float* c2w   = (float*)(ws + sz_cb16);
    u16*   cpack = (u16*)(ws + sz_cb16 + sz_c2);
    u16*   wpack = (u16*)(ws + sz_cb16 + sz_c2 + sz_cpack);
    int*   toks  = (int*)(ws + sz_cb16 + sz_c2 + sz_cpack + sz_wpack);

    float* tok_out = (float*)d_out;                       // [B,S,L] tokens (as float)
    float* emb_out = tok_out + (size_t)BB * SS * LL;      // [B,S,L,KT]

    k_prep_centers<<<LL * KP, 256, 0, stream>>>(centers, cb16, c2w);

    int nCP = LL * 64 * NDC * 256;   // pairs
    k_pack_centers<<<nCP / 256, 256, 0, stream>>>(cb16, (u32*)cpack);

    int nWP = LL * 32 * NDC * 256;   // pairs
    k_pack_w<<<nWP / 256, 256, 0, stream>>>(W, (u32*)wpack);

    k_tokens<<<LL * BB * NSB, 256, 0, stream>>>(h, cpack, c2w, toks, tok_out);

    k_proj<<<LL * BB * NSB, 256, 0, stream>>>(cb16, wpack, bvec, toks, emb_out);
}
